// MultiHeadSpectralAttention_65575560675809
// MI455X (gfx1250) — matrix operand order repaired and checked by reading, not compiled, not run on hardware
//
#include <hip/hip_runtime.h>
#include <hip/hip_bf16.h>

typedef _Float16 half_t;
typedef __attribute__((ext_vector_type(16))) _Float16 v16h;
typedef __attribute__((ext_vector_type(8)))  _Float16 v8h;
typedef __attribute__((ext_vector_type(8)))  float    v8f;

#define NPIX    4096      // H*W
#define C_DIM   256
#define QKV_DIM 768
#define HDIM    32

__device__ __forceinline__ v8f wmma_f16(v16h a, v16h b, v8f c) {
  return __builtin_amdgcn_wmma_f32_16x16x32_f16(false, a, false, b, (short)0, c,
                                                false, false);
}
__device__ __forceinline__ v16h cat8(v8h lo, v8h hi) {
  return __builtin_shufflevector(lo, hi, 0,1,2,3,4,5,6,7,8,9,10,11,12,13,14,15);
}

// ---------------- f32 -> f16 convert (weights) ----------------
__global__ void cvt_h_k(const float* __restrict__ in, half_t* __restrict__ out, int n) {
  for (int i = blockIdx.x * blockDim.x + threadIdx.x; i < n; i += gridDim.x * blockDim.x)
    out[i] = (half_t)in[i];
}

// ---- GroupNorm: in f32 [B,C,N] (channel-major), out f16 [B,N,C] (channel-fastest)
__global__ __launch_bounds__(256) void groupnorm_k(const float* __restrict__ x,
                                                   half_t* __restrict__ out,
                                                   const float* __restrict__ gamma,
                                                   const float* __restrict__ beta) {
  const int CPG = 8;
  const int CNT = CPG * NPIX;                 // 32768 per group
  int b = blockIdx.x >> 5;
  int g = blockIdx.x & 31;
  int tid = threadIdx.x;
  const float* xp = x + ((size_t)b * C_DIM + g * CPG) * NPIX;
  float s = 0.f, q = 0.f;
  for (int i = tid; i < CNT; i += 256) { float v = xp[i]; s += v; q += v * v; }
  __shared__ float ssum[256], ssq[256];
  __shared__ float smu, srs;
  ssum[tid] = s; ssq[tid] = q;
  __syncthreads();
  for (int st = 128; st > 0; st >>= 1) {
    if (tid < st) { ssum[tid] += ssum[tid + st]; ssq[tid] += ssq[tid + st]; }
    __syncthreads();
  }
  if (tid == 0) {
    float mu = ssum[0] / (float)CNT;
    float var = ssq[0] / (float)CNT - mu * mu;
    smu = mu; srs = rsqrtf(var + 1e-5f);
  }
  __syncthreads();
  float mu = smu, rs = srs;
  half_t* op = out + (size_t)b * NPIX * C_DIM;
  for (int i = tid; i < CNT; i += 256) {
    int cl = i >> 12;                          // channel-in-group (coalesced reads)
    int n  = i & 4095;
    float v = (xp[i] - mu) * rs * gamma[g * CPG + cl] + beta[g * CPG + cl];
    op[(size_t)n * C_DIM + g * CPG + cl] = (half_t)v;
  }
}

// ---------------- WMMA GEMM: acc[o,n] = sum_k W[o,k] * Act[n,k] ----------------
// W row-major [O,K]; Act channel-fastest [B,N,K]. Wave tile 16(o) x 64(n).
#define EPI_QKV      0   // split f16 stores: q,k -> [B,N,256]; v -> [B,256,N]
#define EPI_RESID_CN 1   // f32 out[C,N] = resid[C,N] + acc + bias
#define EPI_GELU_NC  2   // f16 out[N,O] = gelu(acc + bias)
__global__ __launch_bounds__(128) void gemm_wmma_k(
    const half_t* __restrict__ W, const half_t* __restrict__ Act,
    float* __restrict__ outF, half_t* __restrict__ outH,
    const float* __restrict__ bias, const float* __restrict__ resid,
    int O, int K, int mode) {
  int wave = threadIdx.x >> 5, lane = threadIdx.x & 31;
  int hi = lane >> 4, l15 = lane & 15;
  int n0 = (blockIdx.x * 4 + wave) * 64;
  int o0 = blockIdx.y * 16;
  int z  = blockIdx.z;
  const half_t* Bp = Act + (size_t)z * NPIX * K;
  const half_t* arow = W + (size_t)(o0 + l15) * K;
  v8f acc0 = {}, acc1 = {}, acc2 = {}, acc3 = {};
  for (int kb = 0; kb < K; kb += 32) {
    v16h a = *(const v16h*)(arow + kb + 16 * hi);
    v16h b0 = *(const v16h*)(Bp + (size_t)(n0      + l15) * K + kb + 16 * hi);
    v16h b1 = *(const v16h*)(Bp + (size_t)(n0 + 16 + l15) * K + kb + 16 * hi);
    v16h b2 = *(const v16h*)(Bp + (size_t)(n0 + 32 + l15) * K + kb + 16 * hi);
    v16h b3 = *(const v16h*)(Bp + (size_t)(n0 + 48 + l15) * K + kb + 16 * hi);
    acc0 = wmma_f16(a, b0, acc0);
    acc1 = wmma_f16(a, b1, acc1);
    acc2 = wmma_f16(a, b2, acc2);
    acc3 = wmma_f16(a, b3, acc3);
  }
  v8f accs[4] = {acc0, acc1, acc2, acc3};
#pragma unroll
  for (int t = 0; t < 4; ++t) {
    int n = n0 + t * 16 + l15;
#pragma unroll
    for (int r = 0; r < 8; ++r) {
      int o = o0 + r + 8 * hi;
      float v = accs[t][r];
      if (mode == EPI_QKV) {
        if (o < 256)
          outH[((size_t)z * NPIX + n) * 256 + o] = (half_t)v;
        else if (o < 512)
          outH[(size_t)2 * NPIX * 256 + ((size_t)z * NPIX + n) * 256 + (o - 256)] = (half_t)v;
        else
          outH[(size_t)4 * NPIX * 256 + ((size_t)z * 256 + (o - 512)) * NPIX + n] = (half_t)v;
      } else if (mode == EPI_RESID_CN) {
        size_t idx = (size_t)z * O * NPIX + (size_t)o * NPIX + n;
        outF[idx] = resid[idx] + v + bias[o];
      } else {
        float tt = v + bias[o];
        outH[(size_t)z * NPIX * O + (size_t)n * O + o] =
            (half_t)(0.5f * tt * (1.0f + erff(tt * 0.70710678118654752f)));
      }
    }
  }
}

// ---------------- flash attention (per-wave, no barriers) ----------------
// q,k f16 [B,N,256] (channel-fastest); v f16 [B,256,N]; out f16 [B,N,256]
__global__ __launch_bounds__(128) void flash_attn_k(
    const half_t* __restrict__ qkv, half_t* __restrict__ attn_out,
    const float* __restrict__ temperature) {
  __shared__ __align__(32) half_t pt[4][16][64];   // per-wave P transpose staging
  int bh = blockIdx.y;
  int b = bh >> 3, h = bh & 7;
  int wave = threadIdx.x >> 5, lane = threadIdx.x & 31;
  int hi = lane >> 4, l15 = lane & 15;
  int i0 = blockIdx.x * 64 + wave * 16;
  const half_t* qp = qkv + ((size_t)b * NPIX) * 256 + h * HDIM;
  const half_t* kp = qkv + (size_t)2 * NPIX * 256 + ((size_t)b * NPIX) * 256 + h * HDIM;
  const half_t* vp = qkv + (size_t)4 * NPIX * 256 + ((size_t)b * 256 + h * HDIM) * NPIX;
  float temp = fminf(fmaxf(temperature[h], 1e-4f), 10.0f);

  // Q A-fragment [16 i x 32 d]: one contiguous 16-element chunk per lane (d = 16*hi + s)
  v16h aq = *(const v16h*)(qp + (size_t)(i0 + l15) * 256 + 16 * hi);
  v8f o0v = {}, o1v = {};
  float mrow[8], lrow[8];
#pragma unroll
  for (int r = 0; r < 8; ++r) { mrow[r] = -3.0e38f; lrow[r] = 0.f; }

  for (int j0 = 0; j0 < NPIX; j0 += 64) {
    // S = Q^T K over a 64-wide j tile: four 16-col B fragments, contiguous v16h
    v16h bk0 = *(const v16h*)(kp + (size_t)(j0      + l15) * 256 + 16 * hi);
    v16h bk1 = *(const v16h*)(kp + (size_t)(j0 + 16 + l15) * 256 + 16 * hi);
    v16h bk2 = *(const v16h*)(kp + (size_t)(j0 + 32 + l15) * 256 + 16 * hi);
    v16h bk3 = *(const v16h*)(kp + (size_t)(j0 + 48 + l15) * 256 + 16 * hi);
    if (j0 + 64 < NPIX)
      __builtin_prefetch(kp + (size_t)(j0 + 64 + l15) * 256, 0, 1);
    v8f zc = {};
    v8f s0 = wmma_f16(aq, bk0, zc);
    v8f s1 = wmma_f16(aq, bk1, zc);
    v8f s2 = wmma_f16(aq, bk2, zc);
    v8f s3 = wmma_f16(aq, bk3, zc);

    // online softmax; row r spans 16 lanes of one half-wave
#pragma unroll
    for (int r = 0; r < 8; ++r) {
      float a0 = s0[r] * temp, a1 = s1[r] * temp;
      float a2 = s2[r] * temp, a3 = s3[r] * temp;
      float rm = fmaxf(fmaxf(a0, a1), fmaxf(a2, a3));
      rm = fmaxf(rm, __shfl_xor(rm, 1));
      rm = fmaxf(rm, __shfl_xor(rm, 2));
      rm = fmaxf(rm, __shfl_xor(rm, 4));
      rm = fmaxf(rm, __shfl_xor(rm, 8));
      float mn = fmaxf(mrow[r], rm);
      float alpha = expf(mrow[r] - mn);
      float p0 = expf(a0 - mn), p1 = expf(a1 - mn);
      float p2 = expf(a2 - mn), p3 = expf(a3 - mn);
      float rs = (p0 + p1) + (p2 + p3);
      rs += __shfl_xor(rs, 1);
      rs += __shfl_xor(rs, 2);
      rs += __shfl_xor(rs, 4);
      rs += __shfl_xor(rs, 8);
      lrow[r] = lrow[r] * alpha + rs;
      mrow[r] = mn;
      o0v[r] *= alpha;
      o1v[r] *= alpha;
      int m = r + 8 * hi;
      pt[wave][m][l15]      = (half_t)p0;
      pt[wave][m][16 + l15] = (half_t)p1;
      pt[wave][m][32 + l15] = (half_t)p2;
      pt[wave][m][48 + l15] = (half_t)p3;
    }

    // P back in A layout (contiguous 16-element loads), V as contiguous B fragments
    v16h ap0 = *(const v16h*)&pt[wave][l15][16 * hi];
    v16h ap1 = *(const v16h*)&pt[wave][l15][32 + 16 * hi];
    v16h bv00 = *(const v16h*)(vp + (size_t)(l15)      * NPIX + j0 + 16 * hi);
    v16h bv01 = *(const v16h*)(vp + (size_t)(l15)      * NPIX + j0 + 32 + 16 * hi);
    v16h bv10 = *(const v16h*)(vp + (size_t)(16 + l15) * NPIX + j0 + 16 * hi);
    v16h bv11 = *(const v16h*)(vp + (size_t)(16 + l15) * NPIX + j0 + 32 + 16 * hi);
    o0v = wmma_f16(ap0, bv00, o0v);
    o0v = wmma_f16(ap1, bv01, o0v);
    o1v = wmma_f16(ap0, bv10, o1v);
    o1v = wmma_f16(ap1, bv11, o1v);
  }

#pragma unroll
  for (int r = 0; r < 8; ++r) {
    float inv = 1.0f / lrow[r];
    int i = i0 + r + 8 * hi;
    attn_out[((size_t)b * NPIX + i) * 256 + h * HDIM + l15]      = (half_t)(o0v[r] * inv);
    attn_out[((size_t)b * NPIX + i) * 256 + h * HDIM + 16 + l15] = (half_t)(o1v[r] * inv);
  }
}

extern "C" void kernel_launch(void* const* d_in, const int* in_sizes, int n_in,
                              void* d_out, int out_size, void* d_ws, size_t ws_size,
                              hipStream_t stream) {
  (void)in_sizes; (void)n_in; (void)out_size; (void)ws_size;
  const float* x      = (const float*)d_in[0];
  const float* w_qkv  = (const float*)d_in[1];
  const float* w_out  = (const float*)d_in[2];
  const float* b_out  = (const float*)d_in[3];
  const float* temp   = (const float*)d_in[4];
  const float* g1     = (const float*)d_in[5];
  const float* beta1  = (const float*)d_in[6];
  const float* g2     = (const float*)d_in[7];
  const float* beta2  = (const float*)d_in[8];
  const float* w_mlp1 = (const float*)d_in[9];
  const float* b_mlp1 = (const float*)d_in[10];
  const float* w_mlp2 = (const float*)d_in[11];
  const float* b_mlp2 = (const float*)d_in[12];
  float* out = (float*)d_out;

  char* ws = (char*)d_ws;
  size_t off = 0;
  auto alloc = [&](size_t bytes) -> char* {
    char* p = ws + off;
    off += (bytes + 255) & ~(size_t)255;
    return p;
  };
  half_t* wq_h  = (half_t*)alloc((size_t)QKV_DIM * C_DIM * sizeof(half_t));
  half_t* wo_h  = (half_t*)alloc((size_t)C_DIM * C_DIM * sizeof(half_t));
  half_t* w1_h  = (half_t*)alloc((size_t)64 * C_DIM * sizeof(half_t));
  half_t* w2_h  = (half_t*)alloc((size_t)C_DIM * 64 * sizeof(half_t));
  half_t* qkv_h = (half_t*)alloc((size_t)6 * NPIX * 256 * sizeof(half_t)); // q|k|v
  half_t* bufB  = (half_t*)alloc((size_t)2 * NPIX * C_DIM * sizeof(half_t)); // xn / y
  half_t* bufC  = (half_t*)alloc((size_t)2 * NPIX * C_DIM * sizeof(half_t)); // attn / mlp-h
  float*  x2    = (float*) alloc((size_t)2 * C_DIM * NPIX * sizeof(float));

  cvt_h_k<<<192, 256, 0, stream>>>(w_qkv, wq_h, QKV_DIM * C_DIM);
  cvt_h_k<<<64,  256, 0, stream>>>(w_out, wo_h, C_DIM * C_DIM);
  cvt_h_k<<<16,  256, 0, stream>>>(w_mlp1, w1_h, 64 * C_DIM);
  cvt_h_k<<<16,  256, 0, stream>>>(w_mlp2, w2_h, C_DIM * 64);

  // GN1: x[C,N] -> xn[N,C] f16
  groupnorm_k<<<64, 256, 0, stream>>>(x, bufB, g1, beta1);
  // qkv: q,k -> [B,N,256]; v -> [B,256,N]
  gemm_wmma_k<<<dim3(16, 48, 2), 128, 0, stream>>>(wq_h, bufB, nullptr, qkv_h,
                                                   nullptr, nullptr, QKV_DIM, C_DIM, EPI_QKV);
  // flash attention -> attn[N,C] f16
  flash_attn_k<<<dim3(64, 16, 1), 128, 0, stream>>>(qkv_h, bufC, temp);
  // x2[C,N] = x + W_out * attn + b_out
  gemm_wmma_k<<<dim3(16, 16, 2), 128, 0, stream>>>(wo_h, bufC, x2, nullptr,
                                                   b_out, x, C_DIM, C_DIM, EPI_RESID_CN);
  // GN2: x2[C,N] -> y[N,C] f16
  groupnorm_k<<<64, 256, 0, stream>>>(x2, bufB, g2, beta2);
  // h[N,64] = gelu(W1 * y + b1)
  gemm_wmma_k<<<dim3(16, 4, 2), 128, 0, stream>>>(w1_h, bufB, nullptr, bufC,
                                                  b_mlp1, nullptr, 64, C_DIM, EPI_GELU_NC);
  // out[C,N] = x2 + W2 * h + b2
  gemm_wmma_k<<<dim3(16, 16, 2), 128, 0, stream>>>(w2_h, bufC, out, nullptr,
                                                   b_mlp2, x2, C_DIM, 64, EPI_RESID_CN);
}
